// LinkWeightDecoder_13142599925966
// MI455X (gfx1250) — matrix operand order repaired and checked by reading, not compiled, not run on hardware
//
#include <hip/hip_runtime.h>

// LinkWeightDecoder for MI455X (gfx1250, wave32, WMMA).
//
// out[e] = relu(relu(concat(emb[src[e]], emb[dst[e]]) @ W1 + b1) @ W2 + b2) @ W3 + b3
//
// emb table (51.2MB) + weights are L2-resident (192MB L2) -> matrix-unit bound.
// Both big GEMMs run on v_wmma_f32_16x16x32_f16 (f16 in, f32 accumulate).
// Pre-pass kernel writes transposed f16 weights to d_ws once; main blocks pull
// them into LDS with async global->LDS b128 copies (ASYNCcnt). Each wave runs
// 4 sixteen-edge tiles with K-outer / N-inner GEMM loops: 8 resident N-tile
// accumulators, single live A fragment (gathered from global with prefetch),
// double-buffered B fragments from LDS -> WMMAs don't stall on dscnt==0.

typedef __attribute__((ext_vector_type(16))) _Float16 v16h;
typedef __attribute__((ext_vector_type(8)))  _Float16 v8h;
typedef __attribute__((ext_vector_type(8)))  float    v8f;
typedef __attribute__((ext_vector_type(4)))  float    v4f;
typedef __attribute__((__vector_size__(4 * sizeof(int)))) int vi4;

#define D_IN  128
#define HID1  128
#define HID2  64
#define THREADS 256            // 8 wave32 waves
#define TILES_PER_WAVE 4
#define EDGES_PER_BLOCK (8 * TILES_PER_WAVE * 16)   // 512

#define W1T_HALVES (HID1 * 2 * D_IN)   // 32768
#define W2T_HALVES (HID2 * HID1)       // 8192
#define WTOTAL_HALVES (W1T_HALVES + W2T_HALVES)     // 40960 -> 80 KB f16

#if defined(__has_builtin)
#  if __has_builtin(__builtin_amdgcn_global_load_async_to_lds_b128)
#    define HAVE_ASYNC_LDS 1
#  endif
#  if __has_builtin(__builtin_amdgcn_s_wait_asynccnt)
#    define HAVE_WAIT_ASYNC_BUILTIN 1
#  endif
#endif

// 16-bit A-matrix 16x32 fragment from an LDS row-major tile.
// Lane L holds row M=L&15; K chunks {k0 + h*8 .. +7} and {k0 + 16 + h*8 .. +7}, h=L>>4.
__device__ __forceinline__ v16h lds_fragA(const _Float16* rowbase, int k0, int h) {
  v8h lo = *(const v8h*)(rowbase + k0 + h * 8);
  v8h hi = *(const v8h*)(rowbase + k0 + 16 + h * 8);
  return __builtin_shufflevector(lo, hi, 0,1,2,3,4,5,6,7,8,9,10,11,12,13,14,15);
}

// 16-bit B-matrix 32x16 fragment from LDS holding W^T row-major ([n][k]).
// Lane L holds column N=L&15; K chunks {k0 + h*8 .. +7} and {k0 + 16 + h*8 .. +7}, h=L>>4.
__device__ __forceinline__ v16h lds_fragB(const _Float16* colbase, int k0, int h) {
  v8h lo = *(const v8h*)(colbase + k0 + h * 8);
  v8h hi = *(const v8h*)(colbase + k0 + 16 + h * 8);
  return __builtin_shufflevector(lo, hi, 0,1,2,3,4,5,6,7,8,9,10,11,12,13,14,15);
}

// Gather one 16x32 f16 A fragment of the concat(src,dst) edge-feature matrix
// straight from the (L2-resident) f32 embedding table. kk 0..3 -> src K-chunks,
// kk 4..7 -> dst K-chunks.
__device__ __forceinline__ v16h gather_fragA(const float* __restrict__ emb,
                                             int myidx, int kk, int h, int l15) {
  const int row = __shfl(myidx, ((kk < 4) ? 0 : 16) + l15, 32);  // ds_bpermute
  const float* p = emb + (long long)row * D_IN + (kk & 3) * 32 + h * 8;
  v4f f0 = *(const v4f*)(p);
  v4f f1 = *(const v4f*)(p + 4);
  v4f g0 = *(const v4f*)(p + 16);
  v4f g1 = *(const v4f*)(p + 20);
  v16h a;
#pragma unroll
  for (int i = 0; i < 4; ++i) {
    a[i]      = (_Float16)f0[i];
    a[4 + i]  = (_Float16)f1[i];
    a[8 + i]  = (_Float16)g0[i];
    a[12 + i] = (_Float16)g1[i];
  }
  return a;
}

// ---- pre-pass: transpose + convert weights into d_ws (f16) ----
// wbuf[0..32767]      = W1^T : wbuf[n*256 + k] = W1[k*128 + n]
// wbuf[32768..40959]  = W2^T : wbuf[32768 + n*128 + k] = W2[k*64 + n]
__global__ __launch_bounds__(THREADS)
void stage_weights_kernel(const float* __restrict__ W1,
                          const float* __restrict__ W2,
                          _Float16* __restrict__ wbuf)
{
  int i = blockIdx.x * THREADS + threadIdx.x;
  if (i < W1T_HALVES) {
    int n = i >> 8, k = i & 255;
    wbuf[i] = (_Float16)W1[k * HID1 + n];
  } else if (i < WTOTAL_HALVES) {
    int j = i - W1T_HALVES;
    int n = j >> 7, k = j & 127;
    wbuf[i] = (_Float16)W2[k * HID2 + n];
  }
}

__global__ __launch_bounds__(THREADS)
void link_decoder_kernel(const float* __restrict__ emb,
                         const long long* __restrict__ ei,
                         const float* __restrict__ W1,
                         const float* __restrict__ b1,
                         const float* __restrict__ W2,
                         const float* __restrict__ b2,
                         const float* __restrict__ W3,
                         const float* __restrict__ b3,
                         const _Float16* __restrict__ wbuf,  // pre-transposed f16
                         int use_ws,
                         float* __restrict__ out,
                         int E)
{
  __shared__ alignas(16) _Float16 wlds[WTOTAL_HALVES];   // 80 KB: w1t | w2t
  __shared__ alignas(16) _Float16 hst[8][16 * HID1];     // 32 KB per-wave h1 staging

  const int tid  = threadIdx.x;
  const int lane = tid & 31;
  const int wave = tid >> 5;
  const int h    = lane >> 4;
  const int l15  = lane & 15;

  _Float16* w1t = wlds;
  _Float16* w2t = wlds + W1T_HALVES;

  // ---- stage weights into LDS ----
  if (use_ws) {
#if HAVE_ASYNC_LDS
    {
      char* gsrc = (char*)(void*)wbuf;
      char* ldst = (char*)(void*)wlds;
      for (int i = tid; i < WTOTAL_HALVES / 8; i += THREADS)
        __builtin_amdgcn_global_load_async_to_lds_b128(
            (vi4*)(gsrc + i * 16), (vi4*)(ldst + i * 16), 0, 0);
#if HAVE_WAIT_ASYNC_BUILTIN
      __builtin_amdgcn_s_wait_asynccnt(0);
#else
      asm volatile("s_wait_asynccnt 0" ::: "memory");
#endif
    }
#else
    {
      const v8h* src = (const v8h*)wbuf;
      v8h* dst = (v8h*)wlds;
      for (int i = tid; i < WTOTAL_HALVES / 8; i += THREADS)
        dst[i] = src[i];
    }
#endif
  } else {
    // fallback: in-kernel transpose+convert from f32 weights
    for (int i = tid; i < W1T_HALVES; i += THREADS) {
      int n = i >> 8, k = i & 255;
      w1t[i] = (_Float16)W1[k * HID1 + n];
    }
    for (int i = tid; i < W2T_HALVES; i += THREADS) {
      int n = i >> 7, k = i & 127;
      w2t[i] = (_Float16)W2[k * HID2 + n];
    }
  }
  __syncthreads();

  // ---- hoist per-lane constants out of the tile loop ----
  float bias1v[8], bias2v[4], w3v[4];
#pragma unroll
  for (int n = 0; n < 8; ++n) bias1v[n] = b1[n * 16 + l15];
#pragma unroll
  for (int n = 0; n < 4; ++n) bias2v[n] = b2[n * 16 + l15];
#pragma unroll
  for (int n = 0; n < 4; ++n) w3v[n] = W3[n * 16 + l15];
  const float b3v = b3[0];

  const long long blockbase = (long long)blockIdx.x * EDGES_PER_BLOCK;
  _Float16* hrow = &hst[wave][0];

  for (int t = 0; t < TILES_PER_WAVE; ++t) {
    const long long ebase = blockbase + (long long)(wave * TILES_PER_WAVE + t) * 16;
    long long eg = ebase + l15;
    if (eg >= E) eg = E - 1;                   // clamp; stores guarded below
    // lanes 0..15: src indices for rows 0..15; lanes 16..31: dst indices
    const int myidx = (int)((lane < 16) ? ei[eg] : ei[(long long)E + eg]);

    // ---- GEMM1: [16 x 256] x [256 x 128] + b1, ReLU -> hst : 64 WMMAs ----
    // K outer / N inner: 8 resident accumulators, A gathered with 1-deep
    // prefetch, B double-buffered from LDS.
    v8f acc1[8];
#pragma unroll
    for (int n = 0; n < 8; ++n) {
      const float b = bias1v[n];
      acc1[n] = (v8f){b, b, b, b, b, b, b, b};
    }

    v16h acur = gather_fragA(emb, myidx, 0, h, l15);
#pragma unroll
    for (int kk = 0; kk < 8; ++kk) {
      v16h anext = acur;
      if (kk < 7) anext = gather_fragA(emb, myidx, kk + 1, h, l15);
      v16h bcur = lds_fragB(w1t + l15 * (2 * D_IN), kk * 32, h);   // n = 0
#pragma unroll
      for (int n = 0; n < 8; ++n) {
        v16h bnext = bcur;
        if (n < 7)
          bnext = lds_fragB(w1t + ((n + 1) * 16 + l15) * (2 * D_IN), kk * 32, h);
        acc1[n] = __builtin_amdgcn_wmma_f32_16x16x32_f16(false, acur, false, bcur,
                                                         (short)0, acc1[n], false, false);
        bcur = bnext;
      }
      acur = anext;
    }

    // ReLU + f16 store. C/D layout: VGPR r, lanes 0-15 -> M=r, 16-31 -> M=r+8.
#pragma unroll
    for (int n = 0; n < 8; ++n) {
#pragma unroll
      for (int r = 0; r < 8; ++r) {
        hrow[(r + h * 8) * HID1 + n * 16 + l15] =
            (_Float16)__builtin_fmaxf(acc1[n][r], 0.f);
      }
    }

    // D-layout -> A-layout transpose stays wave-private: LDS ops from one wave
    // complete in order, so a dscnt drain suffices (no block barrier needed).
    asm volatile("s_wait_dscnt 0" ::: "memory");

    // ---- GEMM2: [16 x 128] x [128 x 64] + b2, ReLU : 16 WMMAs ----
    v8f acc2[4];
#pragma unroll
    for (int n = 0; n < 4; ++n) {
      const float b = bias2v[n];
      acc2[n] = (v8f){b, b, b, b, b, b, b, b};
    }

    const _Float16* hbase = hrow + l15 * HID1;
    v16h hcur = lds_fragA(hbase, 0, h);
#pragma unroll
    for (int kk = 0; kk < 4; ++kk) {
      v16h hnext = hcur;
      if (kk < 3) hnext = lds_fragA(hbase, (kk + 1) * 32, h);
      v16h bcur = lds_fragB(w2t + l15 * HID1, kk * 32, h);         // n = 0
#pragma unroll
      for (int n = 0; n < 4; ++n) {
        v16h bnext = bcur;
        if (n < 3)
          bnext = lds_fragB(w2t + ((n + 1) * 16 + l15) * HID1, kk * 32, h);
        acc2[n] = __builtin_amdgcn_wmma_f32_16x16x32_f16(false, hcur, false, bcur,
                                                         (short)0, acc2[n], false, false);
        bcur = bnext;
      }
      hcur = hnext;
    }

    // ---- ReLU + fused dot with W3 ----
    float rowacc[8];
#pragma unroll
    for (int r = 0; r < 8; ++r) rowacc[r] = 0.f;
#pragma unroll
    for (int n = 0; n < 4; ++n) {
#pragma unroll
      for (int r = 0; r < 8; ++r)
        rowacc[r] += __builtin_fmaxf(acc2[n][r], 0.f) * w3v[n];
    }

    // ---- reduce across each 16-lane half; lanes 0 / 16 hold rows r / r+8 ----
#pragma unroll
    for (int r = 0; r < 8; ++r) {
      float v = rowacc[r];
      v += __shfl_xor(v, 1, 32);
      v += __shfl_xor(v, 2, 32);
      v += __shfl_xor(v, 4, 32);
      v += __shfl_xor(v, 8, 32);
      const long long e = ebase + (r + h * 8);
      if (l15 == 0 && e < E) out[e] = v + b3v;
    }
  }
}

extern "C" void kernel_launch(void* const* d_in, const int* in_sizes, int n_in,
                              void* d_out, int out_size, void* d_ws, size_t ws_size,
                              hipStream_t stream) {
  const float*     emb = (const float*)d_in[0];
  const long long* ei  = (const long long*)d_in[1];   // int64 [2, E]
  const float*     W1  = (const float*)d_in[2];
  const float*     b1  = (const float*)d_in[3];
  const float*     W2  = (const float*)d_in[4];
  const float*     b2  = (const float*)d_in[5];
  const float*     W3  = (const float*)d_in[6];
  const float*     b3  = (const float*)d_in[7];
  float* out = (float*)d_out;

  const int E = in_sizes[1] / 2;
  const int use_ws = (ws_size >= (size_t)WTOTAL_HALVES * sizeof(_Float16)) ? 1 : 0;
  _Float16* wbuf = (_Float16*)d_ws;

  if (use_ws) {
    const int sgrid = (WTOTAL_HALVES + THREADS - 1) / THREADS;   // 160 blocks
    hipLaunchKernelGGL(stage_weights_kernel, dim3(sgrid), dim3(THREADS), 0, stream,
                       W1, W2, wbuf);
  }

  const int grid = (E + EDGES_PER_BLOCK - 1) / EDGES_PER_BLOCK;
  hipLaunchKernelGGL(link_decoder_kernel, dim3(grid), dim3(THREADS), 0, stream,
                     emb, ei, W1, b1, W2, b2, W3, b3, wbuf, use_ws, out, E);
}
